// Conformer_55336358641784
// MI455X (gfx1250) — matrix operand order repaired and checked by reading, not compiled, not run on hardware
//
#include <hip/hip_runtime.h>
#include <hip/hip_bf16.h>

typedef __attribute__((ext_vector_type(16))) __bf16 v16bf;
typedef __attribute__((ext_vector_type(8)))  float  v8f;

// ---------------- problem constants ----------------
constexpr int Bq = 4, Tq = 2048, Dq = 256, Hq = 4, DHq = 64;
constexpr int BT    = Bq * Tq;     // 8192 rows
constexpr int FFIq  = 1024;
constexpr int INNERq = 512;
constexpr float SLOPE = 0.3f;
constexpr float NEGV  = -1e9f;

__device__ __forceinline__ float leakyf(float x){ return x >= 0.f ? x : SLOPE * x; }

// =====================================================================
// bf16 WMMA GEMM:  C[M,N] = A[M,K] * B[N,K]^T  (+ fused epilogue)
//   A: bf16 row-major, leading dim lda
//   B: bf16 [N,K] row-major, leading dim ldb   (all weights pre-transposed)
//   epilogue: +bias[col], optional leaky, optional  v*oscale + res[row,col]
//   stores:   outF (f32) and/or outB (bf16)
// Double-buffered LDS, one barrier per K-step; global loads for step k+1
// are issued before the WMMAs of step k and only stored to LDS afterwards.
// Requires: M%128==0, N%64==0, K%32==0 (true for every call site here).
// =====================================================================
constexpr int BM = 128, BN = 64, BK = 32, LDT = BK + 8; // pad -> conflict-free frag reads

__global__ __launch_bounds__(256) void k_gemm_wmma(
    const __bf16* __restrict__ A, int lda,
    const __bf16* __restrict__ B, int ldb,
    const float* __restrict__ bias, int act,
    const float* __restrict__ res, int ldr, float oscale,
    float* __restrict__ outF, int ldcf,
    __bf16* __restrict__ outB, int ldcb,
    int M, int N, int K)
{
  __shared__ __bf16 As[2][BM * LDT];   // 2 x 10 KB
  __shared__ __bf16 Bs[2][BN * LDT];   // 2 x  5 KB   (stored [n][k])

  const int tid  = threadIdx.x;
  const int m0   = blockIdx.y * BM;
  const int n0   = blockIdx.x * BN;
  const int lane = tid & 31, wave = tid >> 5;
  const int wm   = wave & 3;        // 4 waves along M (32 rows each)
  const int wn   = wave >> 2;       // 2 waves along N (32 cols each)
  const int lr   = lane & 15, lh = lane >> 4;

  // ---- per-thread staging addresses (16B chunks, fully coalesced) ----
  const int ar0 = tid >> 2;              // rows 0..63   of A tile / all of B tile
  const int ar1 = 64 + ar0;              // rows 64..127 of A tile
  const int akc = (tid & 3) * 8;         // k-chunk within 32
  const __bf16* ap0 = A + (size_t)(m0 + ar0) * lda + akc;
  const __bf16* ap1 = A + (size_t)(m0 + ar1) * lda + akc;
  const __bf16* bp  = B + (size_t)(n0 + ar0) * ldb + akc;
  const int sa0 = ar0 * LDT + akc;
  const int sa1 = ar1 * LDT + akc;

  // ---- per-wave fragment offsets (ISA 7.12.2 16-bit layouts) ----
  // A 16x32: lane<16 row=lr K[0..7]+[16..23]; lane>=16 K[8..15]+[24..31]
  const int aoff0 = (wm * 32 +  0 + lr) * LDT + lh * 8;
  const int aoff1 = (wm * 32 + 16 + lr) * LDT + lh * 8;
  // B 32x16: lane<16 col=lr K[0..7]+[16..23]; lane>=16 K[8..15]+[24..31]
  const int boff0 = (wn * 32 +  0 + lr) * LDT + lh * 8;
  const int boff1 = (wn * 32 + 16 + lr) * LDT + lh * 8;

  v8f acc[2][2] = {};

  // ---- prologue: stage K-slab 0 into buffer 0 ----
  {
    uint4 r0 = *(const uint4*)ap0;
    uint4 r1 = *(const uint4*)ap1;
    uint4 rb = *(const uint4*)bp;
    *(uint4*)&As[0][sa0] = r0;
    *(uint4*)&As[0][sa1] = r1;
    *(uint4*)&Bs[0][sa0] = rb;
  }

  int cur = 0;
  for (int k0 = 0; k0 < K; k0 += BK, cur ^= 1) {
    __syncthreads();                      // buf[cur] ready; buf[cur^1] free
    const bool nxt = (k0 + BK) < K;
    uint4 r0, r1, rb;
    if (nxt) {                            // issue next slab's global loads now
      r0 = *(const uint4*)(ap0 + (k0 + BK));
      r1 = *(const uint4*)(ap1 + (k0 + BK));
      rb = *(const uint4*)(bp  + (k0 + BK));
      if (k0 + 2 * BK < K) {              // global_prefetch_b8 hint for k+2
        __builtin_prefetch(ap0 + (k0 + 2 * BK), 0, 1);
        __builtin_prefetch(bp  + (k0 + 2 * BK), 0, 1);
      }
    }

    // ---- WMMA on current buffer ----
    union Frag { v16bf v; uint4 q[2]; } af[2], bfr[2];
    {
      const __bf16* pa0 = &As[cur][aoff0];
      const __bf16* pa1 = &As[cur][aoff1];
      af[0].q[0] = *(const uint4*)pa0;  af[0].q[1] = *(const uint4*)(pa0 + 16);
      af[1].q[0] = *(const uint4*)pa1;  af[1].q[1] = *(const uint4*)(pa1 + 16);
      const __bf16* pb0 = &Bs[cur][boff0];
      const __bf16* pb1 = &Bs[cur][boff1];
      bfr[0].q[0] = *(const uint4*)pb0; bfr[0].q[1] = *(const uint4*)(pb0 + 16);
      bfr[1].q[0] = *(const uint4*)pb1; bfr[1].q[1] = *(const uint4*)(pb1 + 16);
    }
    #pragma unroll
    for (int mt = 0; mt < 2; ++mt)
      #pragma unroll
      for (int nt = 0; nt < 2; ++nt)
        acc[mt][nt] = __builtin_amdgcn_wmma_f32_16x16x32_bf16(
            false, af[mt].v, false, bfr[nt].v, (short)0, acc[mt][nt], false, false);

    // ---- store next slab into the other buffer (no barrier needed) ----
    if (nxt) {
      *(uint4*)&As[cur ^ 1][sa0] = r0;
      *(uint4*)&As[cur ^ 1][sa1] = r1;
      *(uint4*)&Bs[cur ^ 1][sa0] = rb;
    }
  }

  // ---- epilogue: C layout -> lane owns col = lr, rows lh*8 + r ----
  #pragma unroll
  for (int mt = 0; mt < 2; ++mt) {
    #pragma unroll
    for (int nt = 0; nt < 2; ++nt) {
      int col  = n0 + wn * 32 + nt * 16 + lr;
      int row0 = m0 + wm * 32 + mt * 16 + lh * 8;
      float bv = bias ? bias[col] : 0.f;
      #pragma unroll
      for (int r = 0; r < 8; ++r) {
        float v = acc[mt][nt][r] + bv;
        if (act) v = leakyf(v);
        int row = row0 + r;
        if (res)  v = v * oscale + res[(size_t)row * ldr + col];
        if (outF) outF[(size_t)row * ldcf + col] = v;
        if (outB) outB[(size_t)row * ldcb + col] = (__bf16)v;
      }
    }
  }
}

// =====================================================================
// Support kernels (bandwidth-trivial: packing, LN, im2col, softmax, ...)
// =====================================================================
__global__ void k_f32_to_bf16(const float* __restrict__ in, __bf16* __restrict__ out, int n){
  int i = blockIdx.x * 256 + threadIdx.x;
  if (i < n) out[i] = (__bf16)in[i];
}

// transpose-convert: in f32 [K,N] -> out bf16 [N,K]
__global__ void k_convT(const float* __restrict__ in, __bf16* __restrict__ out, int K, int N){
  int i = blockIdx.x * 256 + threadIdx.x;
  if (i >= N * K) return;
  int n = i / K, k = i % K;
  out[i] = (__bf16)in[k * N + n];
}

// ff_w1 [1024,256,3] -> Wff1 [N=1024, K=768] with k = j*256 + c
__global__ void k_ff1_pack(const float* __restrict__ w, __bf16* __restrict__ out, int n){
  int i = blockIdx.x * 256 + threadIdx.x;
  if (i >= n) return;
  int nn = i / 768, k = i % 768;
  int c = k & 255, j = k >> 8;
  out[i] = (__bf16)w[nn * 768 + c * 3 + j];
}

// v [B,T,H*DH] bf16 -> vt [B,H,DH,T] bf16 (per-head [N=DH, K=T] operand)
__global__ void k_vtrans(const __bf16* __restrict__ v, __bf16* __restrict__ vt, int n){
  int i = blockIdx.x * 256 + threadIdx.x;
  if (i >= n) return;
  int s  = i & 2047;
  int r  = i >> 11;
  int dh = r & 63;
  int bh = r >> 6;           // b*4 + h
  int b  = bh >> 2, h = bh & 3;
  vt[i] = v[((size_t)((b << 11) + s)) * 256 + (h << 6) + dh];
}

__global__ __launch_bounds__(256) void k_layernorm(const float* __restrict__ x,
    const float* __restrict__ g, const float* __restrict__ b, __bf16* __restrict__ out){
  __shared__ float s1[256], s2[256];
  int row = blockIdx.x, tid = threadIdx.x;
  float v = x[row * Dq + tid];
  s1[tid] = v; s2[tid] = v * v;
  __syncthreads();
  for (int o = 128; o > 0; o >>= 1){
    if (tid < o){ s1[tid] += s1[tid + o]; s2[tid] += s2[tid + o]; }
    __syncthreads();
  }
  float m   = s1[0] * (1.f / Dq);
  float var = s2[0] * (1.f / Dq) - m * m;
  float r   = rsqrtf(var + 1e-5f);
  out[row * Dq + tid] = (__bf16)((v - m) * r * g[tid] + b[tid]);
}

// im2col for K=3 time-conv, zero padded
__global__ void k_im2col(const __bf16* __restrict__ xb, __bf16* __restrict__ out, int n){
  int i = blockIdx.x * 256 + threadIdx.x;
  if (i >= n) return;
  int k = i % 768, bt = i / 768;
  int j = k >> 8, c = k & 255;
  int b = bt >> 11, t = bt & 2047;
  int tt = t + j - 1;
  __bf16 v = (__bf16)0.f;
  if (tt >= 0 && tt < Tq) v = xb[(b * Tq + tt) * Dq + c];
  out[i] = v;
}

__global__ void k_quqv(const __bf16* __restrict__ q, const float* __restrict__ ub,
    const float* __restrict__ vb, __bf16* __restrict__ qu, __bf16* __restrict__ qv, int n){
  int i = blockIdx.x * 256 + threadIdx.x;
  if (i >= n) return;
  int c = i & 255;
  float f = (float)q[i];
  qu[i] = (__bf16)(f + ub[c]);
  qv[i] = (__bf16)(f + vb[c]);
}

// exact Transformer-XL rel-shift gather + scale + mask + row softmax -> bf16
__global__ __launch_bounds__(256) void k_softmax_shift(const float* __restrict__ S,
    const float* __restrict__ P, const unsigned char* __restrict__ mask,
    __bf16* __restrict__ attn){
  __shared__ float red[256];
  int t = blockIdx.x, tid = threadIdx.x;
  float v[8];
  float mx = -3.4e38f;
  #pragma unroll
  for (int j = 0; j < 8; ++j){
    int s = j * 256 + tid;
    int d = s - t;
    float pv;
    if (d <= 0)      pv = P[t * Tq + (Tq - 1 + d)];
    else if (d == 1) pv = 0.f;
    else             pv = P[(t + 1) * Tq + (d - 2)];   // reference's wrap rows
    float val = (S[t * Tq + s] + pv) * 0.0625f;        // 1/sqrt(D), D=256
    if (mask[s]) val = NEGV;
    v[j] = val;
    mx = fmaxf(mx, val);
  }
  red[tid] = mx; __syncthreads();
  for (int o = 128; o > 0; o >>= 1){ if (tid < o) red[tid] = fmaxf(red[tid], red[tid + o]); __syncthreads(); }
  mx = red[0];
  __syncthreads();
  float sum = 0.f;
  #pragma unroll
  for (int j = 0; j < 8; ++j){ v[j] = __expf(v[j] - mx); sum += v[j]; }
  red[tid] = sum; __syncthreads();
  for (int o = 128; o > 0; o >>= 1){ if (tid < o) red[tid] += red[tid + o]; __syncthreads(); }
  float inv = 1.f / red[0];
  #pragma unroll
  for (int j = 0; j < 8; ++j) attn[t * Tq + j * 256 + tid] = (__bf16)(v[j] * inv);
}

__global__ void k_glu(const float* __restrict__ h, float* __restrict__ u, int n){
  int i = blockIdx.x * 256 + threadIdx.x;
  if (i >= n) return;
  int row = i >> 9, c = i & 511;
  float a = h[row * 1024 + c];
  float g = h[row * 1024 + 512 + c];
  u[i] = a * leakyf(g);
}

__global__ void k_dwconv(const float* __restrict__ u, const float* __restrict__ w,
    const float* __restrict__ bias, float* __restrict__ y, int n){
  int i = blockIdx.x * 256 + threadIdx.x;
  if (i >= n) return;
  int c = i & 511, bt = i >> 9;
  int b = bt >> 11, t = bt & 2047;
  float acc = bias[c];
  #pragma unroll
  for (int j = 0; j < 7; ++j){
    int tt = t + j - 3;
    if (tt >= 0 && tt < Tq) acc += w[c * 7 + j] * u[((b << 11) + tt) * INNERq + c];
  }
  y[i] = acc;
}

// deterministic two-stage GroupNorm(1) reduction over (C,T) per batch
__global__ __launch_bounds__(256) void k_gn_partial(const float* __restrict__ y,
    float* __restrict__ ps1, float* __restrict__ ps2){
  __shared__ float s1[256], s2[256];
  int tid = threadIdx.x;
  int base = blockIdx.x * 256;
  float v = y[base + tid];
  s1[tid] = v; s2[tid] = v * v; __syncthreads();
  for (int o = 128; o > 0; o >>= 1){
    if (tid < o){ s1[tid] += s1[tid + o]; s2[tid] += s2[tid + o]; }
    __syncthreads();
  }
  if (tid == 0){ ps1[blockIdx.x] = s1[0]; ps2[blockIdx.x] = s2[0]; }
}

__global__ __launch_bounds__(256) void k_gn_final(const float* __restrict__ ps1,
    const float* __restrict__ ps2, float* __restrict__ stats){
  __shared__ float s1[256], s2[256];
  int b = blockIdx.x, tid = threadIdx.x;
  const int per = (INNERq * Tq) / 256;   // 4096 partials per batch
  float a = 0.f, c = 0.f;
  for (int i = tid; i < per; i += 256){ a += ps1[b * per + i]; c += ps2[b * per + i]; }
  s1[tid] = a; s2[tid] = c; __syncthreads();
  for (int o = 128; o > 0; o >>= 1){
    if (tid < o){ s1[tid] += s1[tid + o]; s2[tid] += s2[tid + o]; }
    __syncthreads();
  }
  if (tid == 0){ stats[b * 2] = s1[0]; stats[b * 2 + 1] = s2[0]; }
}

__global__ void k_gn_apply(const float* __restrict__ y, const float* __restrict__ stats,
    const float* __restrict__ g, const float* __restrict__ b, __bf16* __restrict__ out, int n){
  int i = blockIdx.x * 256 + threadIdx.x;
  if (i >= n) return;
  int bb = i >> 20, c = i & 511;
  const float invN = 1.f / (float)(INNERq * Tq);
  float m   = stats[bb * 2] * invN;
  float var = stats[bb * 2 + 1] * invN - m * m;
  float r   = rsqrtf(var + 1e-5f);
  out[i] = (__bf16)(leakyf((y[i] - m) * r * g[c] + b[c]));
}

// =====================================================================
// Host orchestration
// =====================================================================
static inline void gemm(hipStream_t st,
    const __bf16* A, int lda, const __bf16* B, int ldb,
    const float* bias, int act, const float* res, int ldr, float oscale,
    float* outF, int ldcf, __bf16* outB, int ldcb, int M, int N, int K){
  dim3 g(N / BN, M / BM);
  k_gemm_wmma<<<g, 256, 0, st>>>(A, lda, B, ldb, bias, act,
                                 res, ldr, oscale, outF, ldcf, outB, ldcb, M, N, K);
}
static inline dim3 ew(long long n){ return dim3((unsigned)((n + 255) / 256)); }

extern "C" void kernel_launch(void* const* d_in, const int* in_sizes, int n_in,
                              void* d_out, int out_size, void* d_ws, size_t ws_size,
                              hipStream_t stream) {
  // ---- inputs (setup_inputs order) ----
  const float* x        = (const float*)d_in[0];
  const float* enc      = (const float*)d_in[1];
  const unsigned char* mask = (const unsigned char*)d_in[2];
  const float* ff_ln_g  = (const float*)d_in[3];
  const float* ff_ln_b  = (const float*)d_in[4];
  const float* ff_w1    = (const float*)d_in[5];
  const float* ff_b1    = (const float*)d_in[6];
  const float* ff_w2    = (const float*)d_in[7];
  const float* ff_b2    = (const float*)d_in[8];
  const float* at_ln_g  = (const float*)d_in[9];
  const float* at_ln_b  = (const float*)d_in[10];
  const float* q_w      = (const float*)d_in[11];
  const float* q_b      = (const float*)d_in[12];
  const float* k_w      = (const float*)d_in[13];
  const float* v_w      = (const float*)d_in[14];
  const float* pos_w    = (const float*)d_in[15];
  const float* u_bias   = (const float*)d_in[16];
  const float* v_bias   = (const float*)d_in[17];
  const float* out_w    = (const float*)d_in[18];
  const float* out_b    = (const float*)d_in[19];
  const float* cm_ln_g  = (const float*)d_in[20];
  const float* cm_ln_b  = (const float*)d_in[21];
  const float* cm_pw1_w = (const float*)d_in[22];
  const float* cm_pw1_b = (const float*)d_in[23];
  const float* cm_dw_w  = (const float*)d_in[24];
  const float* cm_dw_b  = (const float*)d_in[25];
  const float* cm_gn_g  = (const float*)d_in[26];
  const float* cm_gn_b  = (const float*)d_in[27];
  const float* cm_pw2_w = (const float*)d_in[28];
  const float* cm_pw2_b = (const float*)d_in[29];
  float* xout = (float*)d_out;                 // running residual + final output

  // ---- workspace bump allocator ----
  char* base = (char*)d_ws;
  size_t off = 0;
  auto take = [&](size_t bytes) -> char* {
    char* p = base + off;
    off += (bytes + 255) & ~(size_t)255;
    return p;
  };
  __bf16* xb    = (__bf16*)take((size_t)BT * Dq * 2);
  __bf16* encb  = (__bf16*)take((size_t)Tq * Dq * 2);
  __bf16* im2   = (__bf16*)take((size_t)BT * 768 * 2);
  __bf16* h1    = (__bf16*)take((size_t)BT * FFIq * 2);
  __bf16* qb    = (__bf16*)take((size_t)BT * Dq * 2);
  __bf16* qu    = (__bf16*)take((size_t)BT * Dq * 2);
  __bf16* qv    = (__bf16*)take((size_t)BT * Dq * 2);
  __bf16* kb    = (__bf16*)take((size_t)BT * Dq * 2);
  __bf16* vb    = (__bf16*)take((size_t)BT * Dq * 2);
  __bf16* vt    = (__bf16*)take((size_t)BT * Dq * 2);   // [B,H,DH,T]
  __bf16* posb  = (__bf16*)take((size_t)Tq * Dq * 2);
  float*  Sbuf  = (float*)take((size_t)Tq * Tq * 4);
  float*  Pbuf  = (float*)take((size_t)Tq * Tq * 4);
  __bf16* attnb = (__bf16*)take((size_t)Tq * Tq * 2);
  __bf16* ctxb  = (__bf16*)take((size_t)BT * Dq * 2);
  float*  hbig  = (float*)take((size_t)BT * 1024 * 4);
  float*  ubuf  = (float*)take((size_t)BT * 512 * 4);
  float*  y2    = (float*)take((size_t)BT * 512 * 4);
  __bf16* gbf   = (__bf16*)take((size_t)BT * 512 * 2);
  __bf16* wq    = (__bf16*)take((size_t)Dq * Dq * 2);   // [N,K] transposed
  __bf16* wk    = (__bf16*)take((size_t)Dq * Dq * 2);
  __bf16* wv    = (__bf16*)take((size_t)Dq * Dq * 2);
  __bf16* wpos  = (__bf16*)take((size_t)Dq * Dq * 2);
  __bf16* wout  = (__bf16*)take((size_t)Dq * Dq * 2);
  __bf16* wff1  = (__bf16*)take((size_t)1024 * 768 * 2); // [N=1024,K=768]
  __bf16* wff2  = (__bf16*)take((size_t)256 * 1024 * 2); // [N,K] already
  __bf16* wpw1  = (__bf16*)take((size_t)1024 * 256 * 2); // [N,K] already
  __bf16* wpw2  = (__bf16*)take((size_t)256 * 512 * 2);  // [N,K] already
  float*  ps1   = (float*)take((size_t)16384 * 4);
  float*  ps2   = (float*)take((size_t)16384 * 4);
  float*  stats = (float*)take(64);

  // ---- weight / input conversions to bf16 ([N,K] operands) ----
  k_convT<<<ew(65536), 256, 0, stream>>>(q_w,   wq,   256, 256);
  k_convT<<<ew(65536), 256, 0, stream>>>(k_w,   wk,   256, 256);
  k_convT<<<ew(65536), 256, 0, stream>>>(v_w,   wv,   256, 256);
  k_convT<<<ew(65536), 256, 0, stream>>>(pos_w, wpos, 256, 256);
  k_convT<<<ew(65536), 256, 0, stream>>>(out_w, wout, 256, 256);
  k_f32_to_bf16<<<ew(262144), 256, 0, stream>>>(ff_w2,    wff2, 262144);  // [256,1024]
  k_f32_to_bf16<<<ew(262144), 256, 0, stream>>>(cm_pw1_w, wpw1, 262144);  // [1024,256]
  k_f32_to_bf16<<<ew(131072), 256, 0, stream>>>(cm_pw2_w, wpw2, 131072);  // [256,512]
  k_f32_to_bf16<<<ew((size_t)Tq * Dq), 256, 0, stream>>>(enc, encb, Tq * Dq);
  k_ff1_pack<<<ew(786432), 256, 0, stream>>>(ff_w1, wff1, 786432);        // [1024,768]

  // ================= FeedForward (0.5-scaled residual) =================
  k_layernorm<<<BT, 256, 0, stream>>>(x, ff_ln_g, ff_ln_b, xb);
  k_im2col<<<ew((size_t)BT * 768), 256, 0, stream>>>(xb, im2, BT * 768);
  gemm(stream, im2, 768, wff1, 768, ff_b1, 1, nullptr, 0, 0.f,
       nullptr, 0, h1, 1024, BT, 1024, 768);                               // leaky
  gemm(stream, h1, 1024, wff2, 1024, ff_b2, 0, x, 256, 0.5f,
       xout, 256, nullptr, 0, BT, 256, 1024);                              // x + 0.5*y

  // ================= Relative-position MHSA =================
  k_layernorm<<<BT, 256, 0, stream>>>(xout, at_ln_g, at_ln_b, xb);
  gemm(stream, xb, 256, wq, 256, q_b, 0, nullptr, 0, 0.f,
       nullptr, 0, qb, 256, BT, 256, 256);
  gemm(stream, xb, 256, wk, 256, nullptr, 0, nullptr, 0, 0.f,
       nullptr, 0, kb, 256, BT, 256, 256);
  gemm(stream, xb, 256, wv, 256, nullptr, 0, nullptr, 0, 0.f,
       nullptr, 0, vb, 256, BT, 256, 256);
  gemm(stream, encb, 256, wpos, 256, nullptr, 0, nullptr, 0, 0.f,
       nullptr, 0, posb, 256, Tq, 256, 256);
  k_quqv<<<ew((size_t)BT * Dq), 256, 0, stream>>>(qb, u_bias, v_bias, qu, qv, BT * Dq);
  k_vtrans<<<ew((size_t)BT * Dq), 256, 0, stream>>>(vb, vt, BT * Dq);

  for (int b = 0; b < Bq; ++b) {
    for (int h = 0; h < Hq; ++h) {
      size_t obh = (size_t)b * Tq * Dq + (size_t)h * DHq;
      // content scores: (q+u) . k^T          [T,64] x [64,T]
      gemm(stream, qu + obh, 256, kb + obh, 256, nullptr, 0, nullptr, 0, 0.f,
           Sbuf, Tq, nullptr, 0, Tq, Tq, DHq);
      // pos scores (pre-shift): (q+v) . pos^T
      gemm(stream, qv + obh, 256, posb + (size_t)h * DHq, 256, nullptr, 0,
           nullptr, 0, 0.f, Pbuf, Tq, nullptr, 0, Tq, Tq, DHq);
      // rel-shift + scale + mask + softmax -> bf16 probs
      k_softmax_shift<<<Tq, 256, 0, stream>>>(Sbuf, Pbuf, mask + (size_t)b * Tq, attnb);
      // context: attn [T,T] x v^T[DH,T]  (vt is [N=DH, K=T])
      gemm(stream, attnb, Tq, vt + (size_t)(b * Hq + h) * DHq * Tq, Tq,
           nullptr, 0, nullptr, 0, 0.f,
           nullptr, 0, ctxb + obh, 256, Tq, DHq, Tq);
    }
  }
  // out projection + bias + residual
  gemm(stream, ctxb, 256, wout, 256, out_b, 0, xout, 256, 1.f,
       xout, 256, nullptr, 0, BT, 256, 256);

  // ================= Conformer conv module =================
  k_layernorm<<<BT, 256, 0, stream>>>(xout, cm_ln_g, cm_ln_b, xb);
  gemm(stream, xb, 256, wpw1, 256, cm_pw1_b, 0, nullptr, 0, 0.f,
       hbig, 1024, nullptr, 0, BT, 1024, 256);                             // pw1
  k_glu<<<ew((size_t)BT * 512), 256, 0, stream>>>(hbig, ubuf, BT * 512);   // GLU(leaky gate)
  k_dwconv<<<ew((size_t)BT * 512), 256, 0, stream>>>(ubuf, cm_dw_w, cm_dw_b, y2, BT * 512);
  k_gn_partial<<<(BT * 512) / 256, 256, 0, stream>>>(y2, ps1, ps2);
  k_gn_final<<<Bq, 256, 0, stream>>>(ps1, ps2, stats);
  k_gn_apply<<<ew((size_t)BT * 512), 256, 0, stream>>>(y2, stats, cm_gn_g, cm_gn_b, gbf, BT * 512);
  gemm(stream, gbf, 512, wpw2, 512, cm_pw2_b, 0, xout, 256, 1.f,
       xout, 256, nullptr, 0, BT, 256, 512);                               // pw2 + residual
}
